// MultiHeadedAttention_6390911337293
// MI455X (gfx1250) — compile-verified, matrix operand order checked by reading, not run on hardware
//
#include <hip/hip_runtime.h>

// ---------------------------------------------------------------------------
// MI455X (gfx1250) multi-headed channel attention.
// f16 WMMA (v_wmma_f32_16x16x32_f16) with fp32 accumulation for all GEMMs.
// ---------------------------------------------------------------------------

typedef _Float16 f16;
typedef __attribute__((ext_vector_type(16))) _Float16     v16h;
typedef __attribute__((ext_vector_type(8)))  float        v8f;
typedef __attribute__((ext_vector_type(4)))  unsigned int v4u;
typedef __attribute__((ext_vector_type(4)))  float        v4f;

#define LDP 40    // LDS k-pitch in f16 elems: 80B rows, 16B aligned, good bank spread
#define OBP 136   // stage-3 output-assembly LDS pitch (272B, 16B aligned)

// Build a 16x(K=32) f16 WMMA fragment from a row-major [row][k] tile.
// Per CDNA5 ISA 16-bit A/B layout: lane L holds row L%16; VGPRs 0-3 hold
// K = lh*8 .. lh*8+7, VGPRs 4-7 hold K = 16+lh*8 .. 16+lh*8+7 (lh = L/16).
// Caller passes p = &tile[row*pitch + lh*8]; we read 16B at p and p+16elems.
__device__ __forceinline__ v16h frag_ld(const f16* p) {
  union { struct { v4u lo; v4u hi; } u; v16h v; } f;
  f.u.lo = *(const v4u*)(p);
  f.u.hi = *(const v4u*)(p + 16);
  return f.v;
}

__device__ __forceinline__ v8f wmma16(v16h a, v16h b, v8f c) {
  return __builtin_amdgcn_wmma_f32_16x16x32_f16(
      false, a, false, b, (short)0, c, false, false);
}

// ---------------------------------------------------------------------------
// C[b] = W @ X[b] + bias.  W: [M,K] fp32 row-major (shared over batch).
// X: [batch][K][N] (fp32 if XF32 else f16).  C: [batch][M][N] (fp32 if OF32
// else f16).  Block tile 128x128, BK=32, 256 threads = 8 waves (2M x 4N).
// ---------------------------------------------------------------------------
template <bool XF32, bool OF32>
__global__ __launch_bounds__(256) void gemm_bias_kernel(
    const float* __restrict__ W, const void* __restrict__ Xv,
    const float* __restrict__ bias, void* __restrict__ Cv,
    int M, int K, int N) {
  __shared__ alignas(16) f16 As[128 * LDP];  // W tile,   [m][k]
  __shared__ alignas(16) f16 Bs[128 * LDP];  // X tile^T, [n][k]

  const int n0  = blockIdx.x * 128;
  const int m0  = blockIdx.y * 128;
  const int bat = blockIdx.z;

  const int t    = threadIdx.x;
  const int wave = t >> 5, lane = t & 31;
  const int r16  = lane & 15, lh = lane >> 4;
  const int wm   = (wave & 1) * 64;   // wave M offset in block
  const int wn   = (wave >> 1) * 32;  // wave N offset in block

  const float* Xf = (const float*)Xv + (size_t)bat * K * N;
  const f16*   Xh = (const f16*)Xv   + (size_t)bat * K * N;

  v8f acc[4][2];
  for (int mt = 0; mt < 4; ++mt)
    for (int nt = 0; nt < 2; ++nt)
      for (int e = 0; e < 8; ++e) acc[mt][nt][e] = 0.0f;

  for (int k0 = 0; k0 < K; k0 += 32) {
    // Stage W tile: 128 rows x 32 k, float4 loads, f16x4 (8B) LDS stores.
    for (int i = 0; i < 4; ++i) {
      int id = t + i * 256;               // 0..1023 float4 slots
      int row = id >> 3, seg = (id & 7) * 4;
      v4f w4 = *(const v4f*)(W + (size_t)(m0 + row) * K + k0 + seg);
      union { unsigned long long u; f16 h[4]; } hv;
      hv.h[0] = (f16)w4.x; hv.h[1] = (f16)w4.y;
      hv.h[2] = (f16)w4.z; hv.h[3] = (f16)w4.w;
      *(unsigned long long*)(&As[row * LDP + seg]) = hv.u;
    }
    // Stage X tile transposed: coalesced along n, scalar transpose into [n][k].
    for (int i = 0; i < 16; ++i) {
      int id = t + i * 256;               // 0..4095
      int kk = id >> 7, nn = id & 127;
      int ng = n0 + nn; if (ng > N - 1) ng = N - 1;   // clamp; tail masked later
      float xv = XF32 ? Xf[(size_t)(k0 + kk) * N + ng]
                      : (float)Xh[(size_t)(k0 + kk) * N + ng];
      Bs[nn * LDP + kk] = (f16)xv;
    }
    __syncthreads();

    v16h af[4], bfr[2];
    for (int mt = 0; mt < 4; ++mt)
      af[mt] = frag_ld(&As[(wm + mt * 16 + r16) * LDP + lh * 8]);
    for (int nt = 0; nt < 2; ++nt)
      bfr[nt] = frag_ld(&Bs[(wn + nt * 16 + r16) * LDP + lh * 8]);
    for (int mt = 0; mt < 4; ++mt)
      for (int nt = 0; nt < 2; ++nt)
        acc[mt][nt] = wmma16(af[mt], bfr[nt], acc[mt][nt]);
    __syncthreads();
  }

  // Epilogue: D layout -> lane holds n-col = lane%16, rows m = r + 8*(lane/16).
  for (int mt = 0; mt < 4; ++mt) {
    for (int nt = 0; nt < 2; ++nt) {
      int ncol = n0 + wn + nt * 16 + r16;
      if (ncol >= N) continue;
      for (int r = 0; r < 8; ++r) {
        int m = m0 + wm + mt * 16 + r + 8 * lh;
        float v = acc[mt][nt][r] + bias[m];
        size_t off = (size_t)bat * M * N + (size_t)m * N + ncol;
        if (OF32) ((float*)Cv)[off] = v;
        else      ((f16*)Cv)[off]   = (f16)v;
      }
    }
  }
}

// ---------------------------------------------------------------------------
// attn[b,hh] = softmax( Kslice^T @ Qslice / 8 ), 64x64 per (b,hh).
// K/Q buffers viewed flat per batch as [1568][512]; head hh = cols hh*64..+64.
// 128 threads = 4 waves; wave w owns i-tile w, accumulates all 4 j-tiles.
// ---------------------------------------------------------------------------
__global__ __launch_bounds__(128) void attn_scores_kernel(
    const f16* __restrict__ Kb, const f16* __restrict__ Qb,
    f16* __restrict__ attn) {
  __shared__ alignas(16) f16 Ks[64 * LDP];   // [c=i][n] (n-chunk of 32)
  __shared__ alignas(16) f16 Qs[64 * LDP];   // [c=j][n]
  __shared__ float Sf[64 * 65];

  const int hh = blockIdx.x, b = blockIdx.y;
  const int t = threadIdx.x;
  const int wave = t >> 5, lane = t & 31;
  const int r16 = lane & 15, lh = lane >> 4;

  const f16* Kp = Kb + (size_t)b * 512 * 1568 + hh * 64;
  const f16* Qp = Qb + (size_t)b * 512 * 1568 + hh * 64;

  v8f acc[4];
  for (int jt = 0; jt < 4; ++jt)
    for (int e = 0; e < 8; ++e) acc[jt][e] = 0.0f;

  for (int n0 = 0; n0 < 1568; n0 += 32) {      // 49 exact chunks
    for (int i = 0; i < 8; ++i) {
      int nn = wave + i * 4;                   // 0..31
      int c2 = lane * 2;                       // 0..62
      union { unsigned int u; f16 h[2]; } kv, qv;
      kv.u = *(const unsigned int*)(Kp + (size_t)(n0 + nn) * 512 + c2);
      qv.u = *(const unsigned int*)(Qp + (size_t)(n0 + nn) * 512 + c2);
      Ks[(c2    ) * LDP + nn] = kv.h[0];
      Ks[(c2 + 1) * LDP + nn] = kv.h[1];
      Qs[(c2    ) * LDP + nn] = qv.h[0];
      Qs[(c2 + 1) * LDP + nn] = qv.h[1];
    }
    __syncthreads();
    v16h a = frag_ld(&Ks[(wave * 16 + r16) * LDP + lh * 8]);
    for (int jt = 0; jt < 4; ++jt) {
      v16h bq = frag_ld(&Qs[(jt * 16 + r16) * LDP + lh * 8]);
      acc[jt] = wmma16(a, bq, acc[jt]);
    }
    __syncthreads();
  }

  for (int jt = 0; jt < 4; ++jt)
    for (int r = 0; r < 8; ++r)
      Sf[(wave * 16 + r + 8 * lh) * 65 + jt * 16 + r16] = acc[jt][r] * 0.125f;
  __syncthreads();

  if (t < 64) {                                // one row per thread
    float mx = Sf[t * 65];
    for (int j = 1; j < 64; ++j) mx = fmaxf(mx, Sf[t * 65 + j]);
    float s = 0.0f;
    for (int j = 0; j < 64; ++j) {
      float e = __expf(Sf[t * 65 + j] - mx);
      Sf[t * 65 + j] = e; s += e;
    }
    float inv = 1.0f / s;
    f16* dst = attn + ((size_t)(b * 8 + hh) * 64 + t) * 64;
    for (int j = 0; j < 64; ++j) dst[j] = (f16)(Sf[t * 65 + j] * inv);
  }
}

// ---------------------------------------------------------------------------
// Out2[n,i] = Vslice @ attn^T ; scatter AttnOut[b, i*8+hh, n] via LDS so the
// global stores are coalesced 64B runs along n.  V fragments load straight
// from global (layout already matches the A-fragment pattern).
// ---------------------------------------------------------------------------
__global__ __launch_bounds__(256) void attn_apply_kernel(
    const f16* __restrict__ Vb, const f16* __restrict__ attn,
    f16* __restrict__ AO) {
  __shared__ alignas(16) f16 Ob[64 * OBP];     // [i][n_local]

  const int n0 = blockIdx.x * 128;
  const int hh = blockIdx.y, b = blockIdx.z;
  const int N = 1568;
  const int t = threadIdx.x;
  const int wave = t >> 5, lane = t & 31;
  const int r16 = lane & 15, lh = lane >> 4;

  const f16* Ap = attn + (size_t)(b * 8 + hh) * 4096;
  v16h bf[2][4];                               // B frags: attn[i][j], row i, j contig
  for (int jc = 0; jc < 2; ++jc)
    for (int it = 0; it < 4; ++it)
      bf[jc][it] = frag_ld(Ap + (it * 16 + r16) * 64 + jc * 32 + lh * 8);

  int nrow = n0 + wave * 16 + r16;
  if (nrow > N - 1) nrow = N - 1;              // tail waves: garbage, masked below
  const f16* pv = Vb + (size_t)b * 512 * N + (size_t)nrow * 512 + hh * 64;
  v16h a0 = frag_ld(pv + lh * 8);              // j = 0..31
  v16h a1 = frag_ld(pv + 32 + lh * 8);         // j = 32..63

  v8f acc[4];
  for (int it = 0; it < 4; ++it)
    for (int e = 0; e < 8; ++e) acc[it][e] = 0.0f;
  for (int it = 0; it < 4; ++it) {
    acc[it] = wmma16(a0, bf[0][it], acc[it]);
    acc[it] = wmma16(a1, bf[1][it], acc[it]);
  }

  for (int it = 0; it < 4; ++it)
    for (int r = 0; r < 8; ++r)
      Ob[(it * 16 + r16) * OBP + wave * 16 + r + 8 * lh] = (f16)acc[it][r];
  __syncthreads();

  int i = t >> 2, seg = (t & 3) * 32;
  int nout = n0 + seg;
  if (nout < N) {                              // n0 mult of 128, N mult of 32
    f16* dst = AO + ((size_t)b * 512 + i * 8 + hh) * N + nout;
    const f16* src = &Ob[i * OBP + seg];
    for (int q = 0; q < 4; ++q)
      *(v4u*)(dst + q * 8) = *(const v4u*)(src + q * 8);
  }
}

// ---------------------------------------------------------------------------
extern "C" void kernel_launch(void* const* d_in, const int* in_sizes, int n_in,
                              void* d_out, int out_size, void* d_ws,
                              size_t ws_size, hipStream_t stream) {
  (void)in_sizes; (void)n_in; (void)out_size; (void)ws_size;
  const float* x       = (const float*)d_in[0];
  const float* query   = (const float*)d_in[1];
  const float* key_w   = (const float*)d_in[2];
  const float* key_b   = (const float*)d_in[3];
  const float* query_w = (const float*)d_in[4];
  const float* query_b = (const float*)d_in[5];
  const float* val_w   = (const float*)d_in[6];
  const float* val_b   = (const float*)d_in[7];
  const float* up_w    = (const float*)d_in[8];
  const float* up_b    = (const float*)d_in[9];

  const int B = 32, Cin = 1024, Cout = 512, N = 1568;

  char* ws = (char*)d_ws;
  const size_t szKQV = (size_t)B * Cout * N * sizeof(f16);  // 51,380,224 B
  f16* Kb   = (f16*)(ws);
  f16* Qb   = (f16*)(ws + 1 * szKQV);
  f16* Vb   = (f16*)(ws + 2 * szKQV);
  f16* AO   = (f16*)(ws + 3 * szKQV);
  f16* attn = (f16*)(ws + 4 * szKQV);           // 2 MB

  dim3 blk(256);
  dim3 gproj((N + 127) / 128, Cout / 128, B);   // (13, 4, 32)
  gemm_bias_kernel<true, false><<<gproj, blk, 0, stream>>>(
      key_w, x, key_b, Kb, Cout, Cin, N);
  gemm_bias_kernel<true, false><<<gproj, blk, 0, stream>>>(
      query_w, query, query_b, Qb, Cout, Cin, N);
  gemm_bias_kernel<true, false><<<gproj, blk, 0, stream>>>(
      val_w, x, val_b, Vb, Cout, Cin, N);

  attn_scores_kernel<<<dim3(8, B), dim3(128), 0, stream>>>(Kb, Qb, attn);
  attn_apply_kernel<<<dim3((N + 127) / 128, 8, B), dim3(256), 0, stream>>>(
      Vb, attn, AO);

  dim3 gup((N + 127) / 128, Cin / 128, B);      // (13, 8, 32)
  gemm_bias_kernel<false, true><<<gup, blk, 0, stream>>>(
      up_w, AO, up_b, (float*)d_out, Cin, Cout, N);
}
